// MambaBackbone_32040456028561
// MI455X (gfx1250) — matrix operand order repaired and checked by reading, not compiled, not run on hardware
//
#include <hip/hip_runtime.h>

// ---------------- model dims ----------------
#define NLAYERS 4
#define DMODEL  2048
#define DINNER  4096          // 2*DMODEL
#define DSTATE  16
#define DCONV   4
#define DTRANK  128           // DMODEL/16
#define PROJDIM 1536
#define BB      2
#define LL      2048
#define NTOK    (BB*LL)       // 4096
#define XDBLW   (DTRANK + 2*DSTATE)   // 160

typedef __attribute__((ext_vector_type(16))) __bf16 v16bf;
typedef __attribute__((ext_vector_type(8)))  __bf16 v8bf;
typedef __attribute__((ext_vector_type(8)))  float  v8f;

// f32 -> bf16 round-to-nearest-even (bit ops; producer epilogues / one-time
// weight conversion only — never in the GEMM inner loop)
static __device__ __forceinline__ __bf16 f2bf(float f) {
    unsigned u = __builtin_bit_cast(unsigned, f);
    u += 0x7FFFu + ((u >> 16) & 1u);
    unsigned short s = (unsigned short)(u >> 16);
    return __builtin_bit_cast(__bf16, s);
}

// A-fragment (16x32 bf16): lane's 16 values = 16 contiguous elems at p
static __device__ __forceinline__ v16bf loadA_frag(const __bf16* __restrict__ p) {
    v8bf lo = *(const v8bf*)p;
    v8bf hi = *(const v8bf*)(p + 8);
    return __builtin_shufflevector(lo, hi, 0,1,2,3,4,5,6,7,8,9,10,11,12,13,14,15);
}

// B-fragment (32x16 bf16, same k order as A): contiguous 16 K-values of one weight row
static __device__ __forceinline__ v16bf loadB_frag(const __bf16* __restrict__ p) {
    v8bf lo = *(const v8bf*)p;
    v8bf hi = *(const v8bf*)(p + 8);
    return __builtin_shufflevector(lo, hi, 0,1,2,3,4,5,6,7,8,9,10,11,12,13,14,15);
}

// Epilogue modes
#define EPI_NONE     0
#define EPI_SOFTPLUS 1   // v = softplus(v + bias[col])
#define EPI_BIAS     2   // v = v + bias[col]
#define EPI_ACCUM    3   // C[r][c] += v   (residual add)

// ---------------------------------------------------------------------------
// GEMM:  C[M=NTOK, N] = A[M, K] * W[N, K]^T   (bf16 in, f32 WMMA accumulate)
// block = 128 threads = 4 waves; each wave -> 32 rows x 64 cols.
// Per K-step: preload 2 A-frags + 4 B-frags (12 in-order b128 loads), then
// issue 8 WMMAs -> compiler can overlap WMMAs with remaining loads via
// partial loadcnt waits. Branch-free body; unconditional speculative prefetch.
// N-tiles clamped for loads (always-valid memory) and guarded only at store.
// ---------------------------------------------------------------------------
__global__ __launch_bounds__(128) void mamba_gemm_wmma(
    const __bf16* __restrict__ A, int lda,
    const __bf16* __restrict__ W, int ldw,
    float* __restrict__ C, int ldc,
    int N, int K, int epi, const float* __restrict__ bias)
{
    const int lane  = threadIdx.x & 31;
    const int wave  = threadIdx.x >> 5;
    const int half  = lane >> 4;       // 0 | 1
    const int lrow  = lane & 15;
    const int tileM  = (blockIdx.y * 4 + wave) * 32;
    const int tileN0 = blockIdx.x * 64;

    const __bf16* arow0 = A + (size_t)(tileM +      lrow) * lda + half * 16;
    const __bf16* arow1 = A + (size_t)(tileM + 16 + lrow) * lda + half * 16;

    const __bf16* wrow[4];
    v8f acc0[4], acc1[4];
#pragma unroll
    for (int nt = 0; nt < 4; ++nt) {
        int n  = tileN0 + nt * 16;
        int nc = (n + 16 <= N) ? n : (N - 16);        // clamp: loads always valid
        wrow[nt] = W + (size_t)(nc + lrow) * ldw + half * 16;
        acc0[nt] = {};
        acc1[nt] = {};
    }

    for (int k0 = 0; k0 < K; k0 += 32) {
        // ---- issue all fragment loads for this K-slab (in-order returns) ----
        v16bf a0 = loadA_frag(arow0 + k0);
        v16bf a1 = loadA_frag(arow1 + k0);
        v16bf b0 = loadB_frag(wrow[0] + k0);
        v16bf b1 = loadB_frag(wrow[1] + k0);
        v16bf b2 = loadB_frag(wrow[2] + k0);
        v16bf b3 = loadB_frag(wrow[3] + k0);
        // speculative prefetch of next K-slab (dropped silently if OOB)
        __builtin_prefetch(arow0 + k0 + 32, 0, 1);
        __builtin_prefetch(arow1 + k0 + 32, 0, 1);
        __builtin_prefetch(wrow[0] + k0 + 32, 0, 1);
        __builtin_prefetch(wrow[2] + k0 + 32, 0, 1);
        // ---- 8 WMMAs; loads for later tiles still in flight ----
        acc0[0] = __builtin_amdgcn_wmma_f32_16x16x32_bf16(false, a0, false, b0, (short)0, acc0[0], false, false);
        acc1[0] = __builtin_amdgcn_wmma_f32_16x16x32_bf16(false, a1, false, b0, (short)0, acc1[0], false, false);
        acc0[1] = __builtin_amdgcn_wmma_f32_16x16x32_bf16(false, a0, false, b1, (short)0, acc0[1], false, false);
        acc1[1] = __builtin_amdgcn_wmma_f32_16x16x32_bf16(false, a1, false, b1, (short)0, acc1[1], false, false);
        acc0[2] = __builtin_amdgcn_wmma_f32_16x16x32_bf16(false, a0, false, b2, (short)0, acc0[2], false, false);
        acc1[2] = __builtin_amdgcn_wmma_f32_16x16x32_bf16(false, a1, false, b2, (short)0, acc1[2], false, false);
        acc0[3] = __builtin_amdgcn_wmma_f32_16x16x32_bf16(false, a0, false, b3, (short)0, acc0[3], false, false);
        acc1[3] = __builtin_amdgcn_wmma_f32_16x16x32_bf16(false, a1, false, b3, (short)0, acc1[3], false, false);
    }

#pragma unroll
    for (int nt = 0; nt < 4; ++nt) {
        int n = tileN0 + nt * 16;
        if (n + 16 > N) continue;                     // store guard only
        int col = n + lrow;
#pragma unroll
        for (int mt = 0; mt < 2; ++mt) {
#pragma unroll
            for (int j = 0; j < 8; ++j) {
                int r = tileM + mt * 16 + j + half * 8;
                float v = (mt == 0) ? acc0[nt][j] : acc1[nt][j];
                size_t o = (size_t)r * ldc + col;
                if (epi == EPI_SOFTPLUS) {
                    float t = v + bias[col];
                    C[o] = (t > 20.f) ? t : log1pf(__expf(t));
                } else if (epi == EPI_BIAS) {
                    C[o] = v + bias[col];
                } else if (epi == EPI_ACCUM) {
                    C[o] += v;
                } else {
                    C[o] = v;
                }
            }
        }
    }
}

// ---------------------------------------------------------------------------
// Bulk f32 -> bf16 convert (weights once per launch; x_dbl per layer)
// ---------------------------------------------------------------------------
__global__ __launch_bounds__(256) void cvt_f32_bf16_kernel(
    const float* __restrict__ x, __bf16* __restrict__ y, long n)
{
    long i = ((long)blockIdx.x * 256 + threadIdx.x) * 8;
    if (i + 8 > n) return;
    float4 a = ((const float4*)(x + i))[0];
    float4 b = ((const float4*)(x + i))[1];
    v8bf o;
    o[0]=f2bf(a.x); o[1]=f2bf(a.y); o[2]=f2bf(a.z); o[3]=f2bf(a.w);
    o[4]=f2bf(b.x); o[5]=f2bf(b.y); o[6]=f2bf(b.z); o[7]=f2bf(b.w);
    *(v8bf*)(y + i) = o;
}

// ---------------------------------------------------------------------------
// Embedding gather: one block per token (f32 residual stream)
// ---------------------------------------------------------------------------
__global__ __launch_bounds__(256) void embed_kernel(
    const int* __restrict__ ids, const float* __restrict__ emb, float* __restrict__ h)
{
    int t  = blockIdx.x;
    int id = ids[t];
    const float4* src = (const float4*)(emb + (size_t)id * DMODEL);
    float4*       dst = (float4*)(h + (size_t)t * DMODEL);
    for (int i = threadIdx.x; i < DMODEL / 4; i += 256) dst[i] = src[i];
}

// ---------------------------------------------------------------------------
// RMSNorm: one block per token; emits bf16 (feeds WMMA GEMMs directly)
// ---------------------------------------------------------------------------
__global__ __launch_bounds__(256) void rmsnorm_kernel(
    const float* __restrict__ x, const float* __restrict__ w, __bf16* __restrict__ out)
{
    __shared__ float red[256];
    int t = blockIdx.x;
    const float* xr = x + (size_t)t * DMODEL;
    float ss = 0.f;
#pragma unroll
    for (int i = 0; i < DMODEL / 256; ++i) {
        float v = xr[threadIdx.x + i * 256];
        ss += v * v;
    }
    red[threadIdx.x] = ss;
    __syncthreads();
    for (int s = 128; s > 0; s >>= 1) {
        if (threadIdx.x < s) red[threadIdx.x] += red[threadIdx.x + s];
        __syncthreads();
    }
    float scale = rsqrtf(red[0] / (float)DMODEL + 1e-5f);
    __bf16* orow = out + (size_t)t * DMODEL;
#pragma unroll
    for (int i = 0; i < DMODEL / 256; ++i) {
        int c = threadIdx.x + i * 256;
        orow[c] = f2bf(xr[c] * scale * w[c]);
    }
}

// ---------------------------------------------------------------------------
// Depthwise causal conv (D_CONV=4) + SiLU.  Emits f32 (scan) and bf16 (GEMM).
// ---------------------------------------------------------------------------
__global__ __launch_bounds__(256) void conv_silu_kernel(
    const float* __restrict__ xz, const float* __restrict__ cw,
    const float* __restrict__ cb, float* __restrict__ xc, __bf16* __restrict__ xcbf)
{
    int gid = blockIdx.x * 256 + threadIdx.x;
    if (gid >= NTOK * DINNER) return;
    int d = gid & (DINNER - 1);
    int t = gid >> 12;            // DINNER = 4096 = 2^12
    int l = t & (LL - 1);
    float acc = cb[d];
#pragma unroll
    for (int j = 0; j < DCONV; ++j) {
        int li = l - (DCONV - 1) + j;
        if (li >= 0)
            acc += cw[d * DCONV + j] *
                   xz[(size_t)(t - (DCONV - 1) + j) * (2 * DINNER) + d];
    }
    float v = acc / (1.f + __expf(-acc));    // silu
    xc  [(size_t)t * DINNER + d] = v;
    xcbf[(size_t)t * DINNER + d] = f2bf(v);
}

// ---------------------------------------------------------------------------
// Selective scan: one thread per (b, d); 16-state recurrence in registers.
// Fuses  y += D*x  and  y *= silu(z);  emits bf16 (feeds out_proj GEMM).
// ---------------------------------------------------------------------------
__global__ __launch_bounds__(256) void scan_kernel(
    const float* __restrict__ dt,   const float* __restrict__ xdbl,
    const float* __restrict__ xc,   const float* __restrict__ xz,
    const float* __restrict__ Alog, const float* __restrict__ Dp,
    __bf16* __restrict__ y)
{
    int idx = blockIdx.x * 256 + threadIdx.x;   // 0 .. BB*DINNER-1
    int b = idx >> 12;
    int d = idx & (DINNER - 1);

    float A[DSTATE], h[DSTATE];
#pragma unroll
    for (int s = 0; s < DSTATE; ++s) {
        A[s] = -__expf(Alog[d * DSTATE + s]);
        h[s] = 0.f;
    }
    float Dd = Dp[d];

    for (int l = 0; l < LL; ++l) {
        size_t t   = (size_t)b * LL + l;
        float dtv  = dt[t * DINNER + d];
        float xv   = xc[t * DINNER + d];
        const float4* Bp = (const float4*)(xdbl + t * XDBLW + DTRANK);
        const float4* Cp = (const float4*)(xdbl + t * XDBLW + DTRANK + DSTATE);
        float dx   = dtv * xv;
        float yacc = 0.f;
#pragma unroll
        for (int q = 0; q < 4; ++q) {
            float4 Bv = Bp[q];
            float4 Cv = Cp[q];
            int s = q * 4;
            h[s+0] = __expf(dtv * A[s+0]) * h[s+0] + dx * Bv.x;  yacc += h[s+0] * Cv.x;
            h[s+1] = __expf(dtv * A[s+1]) * h[s+1] + dx * Bv.y;  yacc += h[s+1] * Cv.y;
            h[s+2] = __expf(dtv * A[s+2]) * h[s+2] + dx * Bv.z;  yacc += h[s+2] * Cv.z;
            h[s+3] = __expf(dtv * A[s+3]) * h[s+3] + dx * Bv.w;  yacc += h[s+3] * Cv.w;
        }
        float yv = yacc + Dd * xv;
        float zv = xz[t * (2 * DINNER) + DINNER + d];
        y[t * DINNER + d] = f2bf(yv * (zv / (1.f + __expf(-zv))));
    }
}

// ---------------------------------------------------------------------------
// Copy attention mask into the tail of d_out (2nd tuple element)
// ---------------------------------------------------------------------------
__global__ __launch_bounds__(256) void mask_kernel(
    const int* __restrict__ m, float* __restrict__ out, int n)
{
    int i = blockIdx.x * 256 + threadIdx.x;
    if (i < n) out[i] = (float)m[i];
}

// ---------------------------------------------------------------------------
extern "C" void kernel_launch(void* const* d_in, const int* in_sizes, int n_in,
                              void* d_out, int out_size, void* d_ws, size_t ws_size,
                              hipStream_t stream)
{
    (void)in_sizes; (void)n_in; (void)out_size; (void)ws_size;

    const int*   ids    = (const int*)  d_in[0];
    const int*   amask  = (const int*)  d_in[1];
    const float* emb    = (const float*)d_in[2];
    const float* norm_w = (const float*)d_in[3];
    const float* inw    = (const float*)d_in[4];
    const float* cw     = (const float*)d_in[5];
    const float* cb     = (const float*)d_in[6];
    const float* xpw    = (const float*)d_in[7];
    const float* dtw    = (const float*)d_in[8];
    const float* dtbias = (const float*)d_in[9];
    const float* Alog   = (const float*)d_in[10];
    const float* Dp     = (const float*)d_in[11];
    const float* outw   = (const float*)d_in[12];
    const float* normf  = (const float*)d_in[13];
    const float* linw   = (const float*)d_in[14];
    const float* linb   = (const float*)d_in[15];

    // ---- workspace carve-up (256B aligned regions) ----
    char*  base = (char*)d_ws;
    size_t off  = 0;
    auto carve = [&](size_t bytes) -> void* {
        void* p = base + off;
        off += (bytes + 255) & ~(size_t)255;
        return p;
    };
    float*  hbuf   = (float*) carve((size_t)NTOK * DMODEL * 4);
    __bf16* xnbf   = (__bf16*)carve((size_t)NTOK * DMODEL * 2);
    float*  xzb    = (float*) carve((size_t)NTOK * 2 * DINNER * 4);
    float*  xcb    = (float*) carve((size_t)NTOK * DINNER * 4);
    __bf16* xcbf   = (__bf16*)carve((size_t)NTOK * DINNER * 2);
    float*  xdbl   = (float*) carve((size_t)NTOK * XDBLW * 4);
    __bf16* xdblbf = (__bf16*)carve((size_t)NTOK * XDBLW * 2);
    float*  dtb    = (float*) carve((size_t)NTOK * DINNER * 4);
    __bf16* ybf    = (__bf16*)carve((size_t)NTOK * DINNER * 2);
    __bf16* inwbf  = (__bf16*)carve((size_t)NLAYERS * 2 * DINNER * DMODEL * 2);
    __bf16* xpwbf  = (__bf16*)carve((size_t)NLAYERS * XDBLW * DINNER * 2);
    __bf16* dtwbf  = (__bf16*)carve((size_t)NLAYERS * DINNER * DTRANK * 2);
    __bf16* outwbf = (__bf16*)carve((size_t)NLAYERS * DMODEL * DINNER * 2);
    __bf16* linwbf = (__bf16*)carve((size_t)PROJDIM * DMODEL * 2);

    auto cvt = [&](const float* src, __bf16* dst, long n) {
        hipLaunchKernelGGL(cvt_f32_bf16_kernel, dim3((unsigned)(n / (256 * 8))),
                           dim3(256), 0, stream, src, dst, n);
    };
    dim3 gblk(128);
    auto gemm = [&](const __bf16* A, int lda, const __bf16* W, int ldw,
                    float* C, int ldc, int N, int K, int epi, const float* bias) {
        dim3 grid((N + 63) / 64, NTOK / 128);
        hipLaunchKernelGGL(mamba_gemm_wmma, grid, gblk, 0, stream,
                           A, lda, W, ldw, C, ldc, N, K, epi, bias);
    };

    // ---- one-time weight conversion to bf16 ----
    cvt(inw,  inwbf,  (long)NLAYERS * 2 * DINNER * DMODEL);
    cvt(xpw,  xpwbf,  (long)NLAYERS * XDBLW * DINNER);
    cvt(dtw,  dtwbf,  (long)NLAYERS * DINNER * DTRANK);
    cvt(outw, outwbf, (long)NLAYERS * DMODEL * DINNER);
    cvt(linw, linwbf, (long)PROJDIM * DMODEL);

    // ---- embedding ----
    hipLaunchKernelGGL(embed_kernel, dim3(NTOK), dim3(256), 0, stream, ids, emb, hbuf);

    for (int l = 0; l < NLAYERS; ++l) {
        const __bf16* inw_l = inwbf  + (size_t)l * 2 * DINNER * DMODEL;
        const __bf16* xpw_l = xpwbf  + (size_t)l * XDBLW * DINNER;
        const __bf16* dtw_l = dtwbf  + (size_t)l * DINNER * DTRANK;
        const __bf16* ow_l  = outwbf + (size_t)l * DMODEL * DINNER;
        const float*  cw_l  = cw  + (size_t)l * DINNER * DCONV;
        const float*  cb_l  = cb  + (size_t)l * DINNER;
        const float*  dtb_l = dtbias + (size_t)l * DINNER;
        const float*  Al_l  = Alog + (size_t)l * DINNER * DSTATE;
        const float*  D_l   = Dp + (size_t)l * DINNER;

        // xn = rmsnorm(h)  (bf16)
        hipLaunchKernelGGL(rmsnorm_kernel, dim3(NTOK), dim3(256), 0, stream,
                           hbuf, norm_w + (size_t)l * DMODEL, xnbf);
        // xz = xn @ in_w^T            [NTOK, 8192] f32
        gemm(xnbf, DMODEL, inw_l, DMODEL, xzb, 2 * DINNER, 2 * DINNER, DMODEL,
             EPI_NONE, nullptr);
        // xc = silu(conv(xi) + cb)    f32 + bf16
        hipLaunchKernelGGL(conv_silu_kernel, dim3((NTOK * DINNER) / 256), dim3(256),
                           0, stream, xzb, cw_l, cb_l, xcb, xcbf);
        // x_dbl = xc @ x_proj^T       [NTOK, 160] f32, then bf16 copy
        gemm(xcbf, DINNER, xpw_l, DINNER, xdbl, XDBLW, XDBLW, DINNER, EPI_NONE, nullptr);
        cvt(xdbl, xdblbf, (long)NTOK * XDBLW);
        // dt = softplus(dt_r @ dt_w^T + dt_b)
        gemm(xdblbf, XDBLW, dtw_l, DTRANK, dtb, DINNER, DINNER, DTRANK,
             EPI_SOFTPLUS, dtb_l);
        // selective scan (+D*x, *silu(z)) -> bf16
        hipLaunchKernelGGL(scan_kernel, dim3((BB * DINNER) / 256), dim3(256), 0, stream,
                           dtb, xdbl, xcb, xzb, Al_l, D_l, ybf);
        // h += y @ out_w^T
        gemm(ybf, DINNER, ow_l, DINNER, hbuf, DMODEL, DMODEL, DINNER,
             EPI_ACCUM, nullptr);
    }

    // ---- final norm + projection to 1536 ----
    hipLaunchKernelGGL(rmsnorm_kernel, dim3(NTOK), dim3(256), 0, stream, hbuf, normf, xnbf);
    gemm(xnbf, DMODEL, linwbf, DMODEL, (float*)d_out, PROJDIM, PROJDIM, DMODEL,
         EPI_BIAS, linb);

    // tuple element 2: attention mask
    hipLaunchKernelGGL(mask_kernel, dim3((NTOK + 255) / 256), dim3(256), 0, stream,
                       amask, (float*)d_out + (size_t)NTOK * PROJDIM, NTOK);
}
